// CausalSelfAttention_16784732193408
// MI455X (gfx1250) — matrix operand order repaired and checked by reading, not compiled, not run on hardware
//
#include <hip/hip_runtime.h>
#include <hip/hip_bf16.h>
#include <math.h>

// ---------------------------------------------------------------------------
// Causal self-attention (B=4, T=2048, C=1024, H=16, HD=64) for gfx1250.
// fp32 in/out; internal math in bf16 WMMA (v_wmma_f32_16x16x32_bf16, f32 acc).
//   1) x -> bf16 ; W_attn,W_proj -> transposed bf16 (N-major, LDS-tiled)
//   2) QKV GEMM (32x64 per wave, ping-pong double-buffered) -> Q, K, V^T
//   3) flash attention per 16-query tile (S^T orientation, pipelined K/V)
//   4) projection GEMM (32x64 per wave, ping-pong double-buffered) -> fp32 out
// ---------------------------------------------------------------------------

typedef __attribute__((ext_vector_type(16))) __bf16 v16bf;
typedef __attribute__((ext_vector_type(8)))  __bf16 v8bf;
typedef __attribute__((ext_vector_type(8)))  float  v8f;

static constexpr int Bn = 4;
static constexpr int T  = 2048;
static constexpr int C  = 1024;
static constexpr int H  = 16;
static constexpr int HD = 64;

__device__ __forceinline__ v8f wmma_bf16(v16bf a, v16bf b, v8f c) {
  return __builtin_amdgcn_wmma_f32_16x16x32_bf16(false, a, false, b, (short)0, c,
                                                 false, false);
}

// A-fragment (16x32 bf16, M x K), row-major source with leading dim ld.
__device__ __forceinline__ v16bf load_A16(const __bf16* base, int ld) {
  const int lane = threadIdx.x & 31;
  const int row  = lane & 15;
  const int kb   = (lane & 16) ? 16 : 0;
  const __bf16* p = base + (size_t)row * ld + kb;
  v8bf lo = *(const v8bf*)p;
  v8bf hi = *(const v8bf*)(p + 8);
  v16bf a;
#pragma unroll
  for (int i = 0; i < 8; ++i) { a[i] = lo[i]; a[i + 8] = hi[i]; }
  return a;
}

// B-fragment (32x16 bf16, K x N) from B^T stored row-major (N rows, K contig).
__device__ __forceinline__ v16bf load_B16(const __bf16* bt, int ld) {
  const int lane = threadIdx.x & 31;
  const int n    = lane & 15;
  const int kb   = (lane & 16) ? 16 : 0;
  return *(const v16bf*)(bt + (size_t)n * ld + kb);
}

// ------------------------- GEMM fragment bundle ----------------------------
struct GemmBufs { v16bf a0, a1, b0, b1, b2, b3; };

__device__ __forceinline__ GemmBufs load_bufs(const __bf16* arow0,
                                              const __bf16* arow1,
                                              const __bf16* wrow, int k) {
  GemmBufs f;
  f.a0 = load_A16(arow0 + k, C);
  f.a1 = load_A16(arow1 + k, C);
  f.b0 = load_B16(wrow + (size_t)0 * 16 * C + k, C);
  f.b1 = load_B16(wrow + (size_t)1 * 16 * C + k, C);
  f.b2 = load_B16(wrow + (size_t)2 * 16 * C + k, C);
  f.b3 = load_B16(wrow + (size_t)3 * 16 * C + k, C);
  return f;
}

__device__ __forceinline__ void mm8(v8f acc[2][4], const GemmBufs& f) {
  acc[0][0] = wmma_bf16(f.a0, f.b0, acc[0][0]);
  acc[1][0] = wmma_bf16(f.a1, f.b0, acc[1][0]);
  acc[0][1] = wmma_bf16(f.a0, f.b1, acc[0][1]);
  acc[1][1] = wmma_bf16(f.a1, f.b1, acc[1][1]);
  acc[0][2] = wmma_bf16(f.a0, f.b2, acc[0][2]);
  acc[1][2] = wmma_bf16(f.a1, f.b2, acc[1][2]);
  acc[0][3] = wmma_bf16(f.a0, f.b3, acc[0][3]);
  acc[1][3] = wmma_bf16(f.a1, f.b3, acc[1][3]);
}

// ------------------------------ converters ---------------------------------
__global__ __launch_bounds__(256) void k_f32_to_bf16(const float* __restrict__ in,
                                                     __bf16* __restrict__ out,
                                                     size_t n) {
  size_t i = (size_t)blockIdx.x * blockDim.x + threadIdx.x;
  if (i < n) out[i] = (__bf16)in[i];
}

// in: [rows][cols] fp32 row-major ; out: [cols][rows] bf16. rows,cols % 32 == 0.
__global__ __launch_bounds__(256) void k_transpose_bf16(const float* __restrict__ in,
                                                        __bf16* __restrict__ out,
                                                        int rows, int cols) {
  __shared__ float tile[32][33];
  const int tx = threadIdx.x & 31;
  const int ty = threadIdx.x >> 5;       // 0..7
  const int n0 = blockIdx.x * 32;        // input col tile
  const int k0 = blockIdx.y * 32;        // input row tile
#pragma unroll
  for (int i = ty; i < 32; i += 8)
    tile[i][tx] = in[(size_t)(k0 + i) * cols + n0 + tx];
  __syncthreads();
#pragma unroll
  for (int i = ty; i < 32; i += 8)
    out[(size_t)(n0 + i) * rows + k0 + tx] = (__bf16)tile[tx][i];
}

// ------------------------------ QKV GEMM -----------------------------------
// M = B*T = 8192, N = 3C = 3072, K = C. One wave: 32x64 tile, ping-pong DB.
// jobs = 256 m-groups * 48 n-groups = 12288 waves -> 1536 blocks of 8 waves.
__global__ __launch_bounds__(256)
void k_gemm_qkv(const __bf16* __restrict__ xb, const __bf16* __restrict__ wta,
                const float* __restrict__ b_attn, __bf16* __restrict__ Qd,
                __bf16* __restrict__ Kd, __bf16* __restrict__ Vt) {
  const int lane = threadIdx.x & 31;
  const int wave = threadIdx.x >> 5;
  const int job  = blockIdx.x * 8 + wave;
  const int NG   = (3 * C) / 64;   // 48
  const int mg   = job / NG;       // 0..255
  const int ng   = job % NG;

  const __bf16* arow0 = xb + (size_t)mg * 32 * C;
  const __bf16* arow1 = arow0 + (size_t)16 * C;
  const __bf16* wrow  = wta + (size_t)(ng * 64) * C;

  v8f zero = {};
  v8f acc[2][4];
#pragma unroll
  for (int g = 0; g < 2; ++g)
#pragma unroll
    for (int j = 0; j < 4; ++j) acc[g][j] = zero;

  GemmBufs fA = load_bufs(arow0, arow1, wrow, 0);
  int k = 32;
  for (; k + 32 < C; k += 64) {
    GemmBufs fB = load_bufs(arow0, arow1, wrow, k);
    mm8(acc, fA);
    fA = load_bufs(arow0, arow1, wrow, k + 32);
    mm8(acc, fB);
  }
  {
    GemmBufs fB = load_bufs(arow0, arow1, wrow, k);  // k == C-32
    mm8(acc, fA);
    mm8(acc, fB);
  }

  const int hi8 = (lane & 16) ? 8 : 0;
  const int qn  = lane & 15;
#pragma unroll
  for (int g = 0; g < 2; ++g) {
#pragma unroll
    for (int j = 0; j < 4; ++j) {
      const int   n    = ng * 64 + j * 16 + qn;
      const float bias = b_attn[n];
      const int part = n >> 10;       // 0:q 1:k 2:v
      const int cc   = n & (C - 1);
      const int h    = cc >> 6;
      const int d    = cc & 63;
#pragma unroll
      for (int r = 0; r < 8; ++r) {
        const int m  = mg * 32 + g * 16 + r + hi8;
        const int bb = m >> 11;       // / T
        const int t  = m & (T - 1);
        const __bf16 v = (__bf16)(acc[g][j][r] + bias);
        const size_t bh = (size_t)bb * H + h;
        if (part == 0)      Qd[(bh * T + t) * HD + d] = v;
        else if (part == 1) Kd[(bh * T + t) * HD + d] = v;
        else                Vt[(bh * HD + d) * T + t] = v;
      }
    }
  }
}

// --------------------------- flash attention -------------------------------
// One wave per (bh, 16-query tile). S^T = K.Q^T orientation: softmax row lives
// in a lane pair (lane, lane^16) -> one __shfl_xor per reduction. O^T = V^T.P.
// K fragments double-buffered across chunks; V^T loads issued before the
// softmax VALU block so they overlap exp/max/shuffle work.
__global__ __launch_bounds__(256)
void k_attention(const __bf16* __restrict__ Qd, const __bf16* __restrict__ Kd,
                 const __bf16* __restrict__ Vt, __bf16* __restrict__ Yb) {
  const int lane = threadIdx.x & 31;
  const int wave = threadIdx.x >> 5;
  const int job  = blockIdx.x * 8 + wave;
  const int qt   = job & (T / 16 - 1);  // 0..127
  const int bh   = job >> 7;            // 0..63
  const int q0   = qt * 16;
  const int qn   = lane & 15;
  const int hi   = (lane & 16) ? 1 : 0;
  const int hi8  = hi << 3;
  const int q    = q0 + qn;

  // Q^T B-fragments (d 0..31 and 32..63), loaded once.
  const int kb16 = hi ? 16 : 0;
  const __bf16* qrow = Qd + ((size_t)bh * T + q0 + qn) * HD;
  v16bf qb0 = *(const v16bf*)(qrow + kb16);
  v16bf qb1 = *(const v16bf*)(qrow + 32 + kb16);

  const __bf16* kbh = Kd + (size_t)bh * T * HD;
  const __bf16* vbh = Vt + (size_t)bh * HD * T;

  v8f zero = {};
  v8f o0 = zero, o1 = zero, o2 = zero, o3 = zero;   // O^T: [64 d x 16 q]
  float mrun = -1e30f, lrun = 0.f;
  const float scale = 0.125f;  // 1/sqrt(64)
  const int cend = q0 + 16;

  // preload K fragments for chunk 0 (keys 0..31, each over d 0..31 / 32..63)
  v16bf ka0 = load_A16(kbh, HD);
  v16bf ka1 = load_A16(kbh + 32, HD);
  v16bf ka2 = load_A16(kbh + 16 * HD, HD);
  v16bf ka3 = load_A16(kbh + 16 * HD + 32, HD);

  for (int c = 0; c < cend; c += 32) {
    v8f s0 = zero, s1 = zero;
    s0 = wmma_bf16(ka0, qb0, s0);
    s0 = wmma_bf16(ka1, qb1, s0);
    s1 = wmma_bf16(ka2, qb0, s1);
    s1 = wmma_bf16(ka3, qb1, s1);

    // issue V^T loads for this chunk now; they complete under the softmax VALU
    const __bf16* vb = vbh + c;
    v16bf va0 = load_A16(vb + (size_t)0  * T, T);
    v16bf va1 = load_A16(vb + (size_t)16 * T, T);
    v16bf va2 = load_A16(vb + (size_t)32 * T, T);
    v16bf va3 = load_A16(vb + (size_t)48 * T, T);

    // prefetch next chunk's K fragments (written directly into loop-carried regs)
    if (c + 32 < cend) {
      const __bf16* kn = kbh + (size_t)(c + 32) * HD;
      ka0 = load_A16(kn, HD);
      ka1 = load_A16(kn + 32, HD);
      ka2 = load_A16(kn + 16 * HD, HD);
      ka3 = load_A16(kn + 16 * HD + 32, HD);
    }

    float e0[8], e1[8];
    float cmax = -1e30f;
#pragma unroll
    for (int r = 0; r < 8; ++r) {
      const int key0 = c + r + hi8;
      const int key1 = key0 + 16;
      float x0 = (key0 <= q) ? s0[r] * scale : -1e30f;
      float x1 = (key1 <= q) ? s1[r] * scale : -1e30f;
      e0[r] = x0; e1[r] = x1;
      cmax = fmaxf(cmax, fmaxf(x0, x1));
    }
    cmax = fmaxf(cmax, __shfl_xor(cmax, 16));
    const float nm   = fmaxf(mrun, cmax);
    const float corr = __expf(mrun - nm);
    mrun = nm;

    float lsum = 0.f;
#pragma unroll
    for (int r = 0; r < 8; ++r) {
      e0[r] = __expf(e0[r] - nm);
      e1[r] = __expf(e1[r] - nm);
      lsum += e0[r] + e1[r];
    }
    lsum += __shfl_xor(lsum, 16);
    lrun = lrun * corr + lsum;
#pragma unroll
    for (int r = 0; r < 8; ++r) { o0[r] *= corr; o1[r] *= corr; o2[r] *= corr; o3[r] *= corr; }

    // Rebuild P as a B-fragment (col = q, 16 contiguous keys per lane):
    // swap 8-key halves between lane and lane^16.
    v16bf pf;
#pragma unroll
    for (int r = 0; r < 8; ++r) {
      float a0 = e0[r], a1 = e1[r];
      float b0 = __shfl_xor(a0, 16);
      float b1 = __shfl_xor(a1, 16);
      pf[r]     = (__bf16)(hi ? b1 : a0);
      pf[r + 8] = (__bf16)(hi ? a1 : b0);
    }

    // O^T += V^T-tile . P
    o0 = wmma_bf16(va0, pf, o0);
    o1 = wmma_bf16(va1, pf, o1);
    o2 = wmma_bf16(va2, pf, o2);
    o3 = wmma_bf16(va3, pf, o3);
  }

  const float linv = 1.f / lrun;
  const int bb = bh >> 4, h = bh & (H - 1);
  __bf16* yrow = Yb + ((size_t)bb * T + q0 + qn) * C + h * HD;
#pragma unroll
  for (int r = 0; r < 8; ++r) {
    yrow[0  + r + hi8] = (__bf16)(o0[r] * linv);
    yrow[16 + r + hi8] = (__bf16)(o1[r] * linv);
    yrow[32 + r + hi8] = (__bf16)(o2[r] * linv);
    yrow[48 + r + hi8] = (__bf16)(o3[r] * linv);
  }
}

// --------------------------- projection GEMM -------------------------------
// M = 8192, N = C, K = C. One wave: 32x64, ping-pong double-buffered.
// jobs = 256 * 16 = 4096 waves -> 512 blocks of 8 waves.
__global__ __launch_bounds__(256)
void k_gemm_proj(const __bf16* __restrict__ yb, const __bf16* __restrict__ wtp,
                 const float* __restrict__ b_proj, float* __restrict__ out) {
  const int lane = threadIdx.x & 31;
  const int wave = threadIdx.x >> 5;
  const int job  = blockIdx.x * 8 + wave;
  const int NG   = C / 64;  // 16
  const int mg   = job / NG;
  const int ng   = job % NG;

  const __bf16* arow0 = yb + (size_t)mg * 32 * C;
  const __bf16* arow1 = arow0 + (size_t)16 * C;
  const __bf16* wrow  = wtp + (size_t)(ng * 64) * C;

  v8f zero = {};
  v8f acc[2][4];
#pragma unroll
  for (int g = 0; g < 2; ++g)
#pragma unroll
    for (int j = 0; j < 4; ++j) acc[g][j] = zero;

  GemmBufs fA = load_bufs(arow0, arow1, wrow, 0);
  int k = 32;
  for (; k + 32 < C; k += 64) {
    GemmBufs fB = load_bufs(arow0, arow1, wrow, k);
    mm8(acc, fA);
    fA = load_bufs(arow0, arow1, wrow, k + 32);
    mm8(acc, fB);
  }
  {
    GemmBufs fB = load_bufs(arow0, arow1, wrow, k);  // k == C-32
    mm8(acc, fA);
    mm8(acc, fB);
  }

  const int hi8 = (lane & 16) ? 8 : 0;
  const int qn  = lane & 15;
#pragma unroll
  for (int g = 0; g < 2; ++g) {
#pragma unroll
    for (int j = 0; j < 4; ++j) {
      const int   n    = ng * 64 + j * 16 + qn;
      const float bias = b_proj[n];
#pragma unroll
      for (int r = 0; r < 8; ++r) {
        const int m = mg * 32 + g * 16 + r + hi8;
        out[(size_t)m * C + n] = acc[g][j][r] + bias;
      }
    }
  }
}

// ------------------------------- launcher ----------------------------------
extern "C" void kernel_launch(void* const* d_in, const int* in_sizes, int n_in,
                              void* d_out, int out_size, void* d_ws, size_t ws_size,
                              hipStream_t stream) {
  (void)in_sizes; (void)n_in; (void)out_size; (void)ws_size;
  const float* x      = (const float*)d_in[0];
  const float* W_attn = (const float*)d_in[1];
  const float* b_attn = (const float*)d_in[2];
  const float* W_proj = (const float*)d_in[3];
  const float* b_proj = (const float*)d_in[4];
  float*       out    = (float*)d_out;

  const size_t BTC = (size_t)Bn * T * C;   // 8M elements
  char* ws = (char*)d_ws;
  size_t off = 0;
  __bf16* xb  = (__bf16*)(ws + off); off += BTC * 2;               // 16 MB
  __bf16* wta = (__bf16*)(ws + off); off += (size_t)3 * C * C * 2; // 6 MB
  __bf16* wtp = (__bf16*)(ws + off); off += (size_t)C * C * 2;     // 2 MB
  __bf16* Qd  = (__bf16*)(ws + off); off += BTC * 2;               // 16 MB
  __bf16* Kd  = (__bf16*)(ws + off); off += BTC * 2;               // 16 MB
  __bf16* Vt  = (__bf16*)(ws + off); off += BTC * 2;               // 16 MB
  __bf16* Yb  = (__bf16*)(ws + off); off += BTC * 2;               // 16 MB

  k_f32_to_bf16<<<(int)((BTC + 255) / 256), 256, 0, stream>>>(x, xb, BTC);
  {
    dim3 g1((3 * C) / 32, C / 32);
    k_transpose_bf16<<<g1, 256, 0, stream>>>(W_attn, wta, C, 3 * C);
    dim3 g2(C / 32, C / 32);
    k_transpose_bf16<<<g2, 256, 0, stream>>>(W_proj, wtp, C, C);
  }
  k_gemm_qkv<<<1536, 256, 0, stream>>>(xb, wta, b_attn, Qd, Kd, Vt);
  k_attention<<<1024, 256, 0, stream>>>(Qd, Kd, Vt, Yb);
  k_gemm_proj<<<512, 256, 0, stream>>>(Yb, wtp, b_proj, out);
}
